// RingGNN_2422361555586
// MI455X (gfx1250) — matrix operand order repaired and checked by reading, not compiled, not run on hardware
//
#include <hip/hip_runtime.h>
#include <math.h>

// ---------------------------------------------------------------------------
// RingGNN forward for MI455X (gfx1250, wave32, WMMA).
// B=8, n=128, HIDDEN=64, HF=128, NC=10, RADIUS=2, NLAYERS=2.
// Internal feature layout: channels-first [B][C][n][n] fp32.
// GEMMs use v_wmma_f32_16x16x32_f16 (f16 in, fp32 accumulate).
// LDS B-tiles are stored column-major (Bt[col][K], stride 40 f16) so each
// B-fragment is two ds_load_b128; A-fragments are two global_load_b128.
// ---------------------------------------------------------------------------

typedef __attribute__((ext_vector_type(16))) _Float16 v16h;
typedef __attribute__((ext_vector_type(8)))  float    v8f;

#define GB 8      // batches
#define GN 128    // graph size n
#define GC 64     // hidden channels
#define GHF 128   // feature width
#define GNC 10    // classes
#define NN (GN*GN)
// Column-major LDS tile: Bt[col*CSTR + K], K = 0..31 per column.
// CSTR = 40 f16 (80 B = 20 dwords) -> 16-lane b128 reads are bank-conflict-free.
#define CSTR 40

union FragH {
  v16h h;
  uint u[8];
  uint4 u4[2];
};

__device__ __forceinline__ v8f wmma_f16(v16h a, v16h b, v8f c) {
  // 8 args: (neg_a, A, neg_b, B, c_mod, C, reuse_a, reuse_b)
  return __builtin_amdgcn_wmma_f32_16x16x32_f16(false, a, false, b, (short)0, c,
                                                false, false);
}

// A-matrix (16x32, f16) fragment, contiguous-K order used here (same as load_b_frag):
//   lane: M = lane&15, half = lane>>4
//   VGPR v0..3 <- K = half*16 + {0..7}; v4..7 <- K = half*16 + 8 + {0..7}
// Both runs are contiguous in row-major memory -> two 16-byte loads.
__device__ __forceinline__ v16h load_a_frag(const _Float16* row, int half) {
  FragH f;
  f.u4[0] = *(const uint4*)(row + half * 16);
  f.u4[1] = *(const uint4*)(row + half * 16 + 8);
  return f.h;
}

// B-matrix (32x16, f16) fragment from the column-major LDS tile:
//   lane: N = col, half = lane>>4 ; VGPR v <- K = half*16 + {2v, 2v+1}
//   => one lane needs K = half*16 .. half*16+15, contiguous at fixed col:
//   two aligned 16-byte LDS loads.
__device__ __forceinline__ v16h load_b_frag(const _Float16* Bt, int col, int half) {
  FragH f;
  const _Float16* p = Bt + col * CSTR + half * 16;
  f.u4[0] = *(const uint4*)(p);
  f.u4[1] = *(const uint4*)(p + 8);
  return f.h;
}

// ---------------------------------------------------------------------------
// Per-(b,c) image statistics: rows[i], cols[j], diag[i], trace, total.
// grid = B*C blocks, 128 threads.
// ---------------------------------------------------------------------------
__global__ void rg_stats_kernel(const float* __restrict__ u,
                                float* __restrict__ rows, float* __restrict__ cols,
                                float* __restrict__ diag, float* __restrict__ sd,
                                float* __restrict__ tot) {
  int bc = blockIdx.x;
  int tid = threadIdx.x;
  const float* img = u + (size_t)bc * NN;
  float rowacc = 0.f, colacc = 0.f;
  for (int j = 0; j < GN; ++j) rowacc += img[tid * GN + j];
  for (int i = 0; i < GN; ++i) colacc += img[i * GN + tid];
  float dv = img[tid * (GN + 1)];
  rows[(size_t)bc * GN + tid] = rowacc;
  cols[(size_t)bc * GN + tid] = colacc;
  diag[(size_t)bc * GN + tid] = dv;
  __shared__ float r1[GN], r2[GN];
  r1[tid] = rowacc; r2[tid] = dv;
  __syncthreads();
  for (int off = 64; off > 0; off >>= 1) {
    if (tid < off) { r1[tid] += r1[tid + off]; r2[tid] += r2[tid + off]; }
    __syncthreads();
  }
  if (tid == 0) { tot[bc] = r1[0]; sd[bc] = r2[0]; }
}

// ---------------------------------------------------------------------------
// Rank-1 terms of the 15-basis map for one coeff set (din x 64 x 15):
//  Di (delta),  Ri (row-broadcast),  Cj (col-broadcast),  Kc (constant).
// grid = B*64 blocks (b,s), 128 threads (i).
// ---------------------------------------------------------------------------
__global__ void rg_vec_kernel(const float* __restrict__ coeffs, int din,
                              const float* __restrict__ rows,
                              const float* __restrict__ cols,
                              const float* __restrict__ diag,
                              const float* __restrict__ sd,
                              const float* __restrict__ tot,
                              float* __restrict__ Ri, float* __restrict__ Cj,
                              float* __restrict__ Di, float* __restrict__ Kc) {
  int blk = blockIdx.x;
  int b = blk >> 6, s = blk & 63;
  int i = threadIdx.x;
  const float inv_n = 1.f / (float)GN, inv_n2 = 1.f / (float)NN;
  float ri = 0.f, cj = 0.f, di = 0.f, kc = 0.f;
  for (int d = 0; d < din; ++d) {
    const float* c = coeffs + ((size_t)d * GC + s) * 15;
    size_t bd = (size_t)b * din + d;
    float rv = rows[bd * GN + i];
    float cv = cols[bd * GN + i];
    float dv = diag[bd * GN + i];
    float sdv = sd[bd], tv = tot[bd];
    di += c[0] * dv + (c[1] * sdv + c[2] * rv + c[3] * cv) * inv_n + c[4] * tv * inv_n2;
    ri += (c[5] * cv + c[6] * rv) * inv_n + c[11] * dv;
    cj += (c[7] * cv + c[8] * rv) * inv_n + c[12] * dv;
    kc += c[13] * sdv * inv_n + c[14] * tv * inv_n2;
  }
  size_t bs = (size_t)b * GC + s;
  Ri[bs * GN + i] = ri;
  Cj[bs * GN + i] = cj;
  Di[bs * GN + i] = di;
  if (i == 0) Kc[bs] = kc;
}

// ---------------------------------------------------------------------------
// Pack A' = [W9 | W10] (64 x 128) to f16 row-major for the channel-mix GEMM.
// ---------------------------------------------------------------------------
__global__ void rg_packA_kernel(const float* __restrict__ coeffs,
                                _Float16* __restrict__ ah) {
  for (int t = threadIdx.x; t < GC * 128; t += 256) {
    int s = t >> 7, k = t & 127;
    float v = (k < GC) ? coeffs[((size_t)k * GC + s) * 15 + 9]
                       : coeffs[((size_t)(k - GC) * GC + s) * 15 + 10];
    ah[t] = (_Float16)v;
  }
}

// ---------------------------------------------------------------------------
// Layer-1 basis map (din=1): pure elementwise.
// grid = B*64 blocks, 256 threads.
// ---------------------------------------------------------------------------
__global__ void rg_mix1_kernel(const float* __restrict__ A,
                               const float* __restrict__ coeffs,
                               const float* __restrict__ Ri, const float* __restrict__ Cj,
                               const float* __restrict__ Di, const float* __restrict__ Kc,
                               const float* __restrict__ dbias,
                               float* __restrict__ outF, _Float16* __restrict__ outH) {
  int blk = blockIdx.x;
  int b = blk >> 6, s = blk & 63;
  float c9 = coeffs[s * 15 + 9], c10 = coeffs[s * 15 + 10];
  size_t bs = (size_t)b * GC + s;
  float kc = Kc[bs];
  const float* Ab = A + (size_t)b * NN;
  for (int t = threadIdx.x; t < NN; t += 256) {
    int i = t >> 7, j = t & 127;
    float val = c9 * Ab[i * GN + j] + c10 * Ab[j * GN + i] +
                Ri[bs * GN + i] + Cj[bs * GN + j] + kc;
    if (i == j) { val += Di[bs * GN + i]; if (dbias) val += dbias[s]; }
    size_t g = bs * NN + t;
    if (outF) outF[g] = val; else outH[g] = (_Float16)val;
  }
}

// ---------------------------------------------------------------------------
// Layer-2 basis map (din=64), WMMA: T = A'(64x128) x [X ; X^T](128x16384) + rank1.
// grid = B*128 blocks (one per (b, output row i)), 256 threads = 8 waves.
// Wave w: M-tile = w&3 (16 channels), N-half = w>>2 (64 of the 128 columns).
// K loop: 4 steps of 32 (k<64 -> X rows, k>=64 -> X^T rows).
// ---------------------------------------------------------------------------
__global__ void rg_mix64_kernel(const float* __restrict__ u,
                                const _Float16* __restrict__ ah,
                                const float* __restrict__ Ri, const float* __restrict__ Cj,
                                const float* __restrict__ Di, const float* __restrict__ Kc,
                                const float* __restrict__ dbias,
                                float* __restrict__ outF, _Float16* __restrict__ outH) {
  __shared__ __align__(16) _Float16 Bt[GN * CSTR];
  int blk = blockIdx.x;
  int b = blk >> 7, i = blk & 127;
  int tid = threadIdx.x, lane = tid & 31, wave = tid >> 5;
  int half = lane >> 4, l15 = lane & 15;
  int mtile = wave & 3, nhalf = wave >> 2;
  v8f acc[4];
#pragma unroll
  for (int t = 0; t < 4; ++t)
#pragma unroll
    for (int v = 0; v < 8; ++v) acc[t][v] = 0.f;

  const float* ub = u + (size_t)b * GC * NN;
  for (int kk = 0; kk < 4; ++kk) {
    // Stage 32x128 B tile column-major: element (dd,j) at Bt[j*CSTR + dd].
    // Each thread converts+packs the K-pair (dd0, dd0+1) for one column.
    for (int t = tid; t < 16 * GN; t += 256) {
      int pr = t >> 7;            // 0..15 -> dd0 = 2*pr
      int j = t & 127;
      int dd0 = pr * 2;
      float x0, x1;
      if (kk < 2) {
        const float* p = ub + ((size_t)(kk * 32 + dd0) * GN + i) * GN + j;
        x0 = p[0]; x1 = p[NN];
      } else {
        const float* p = ub + ((size_t)((kk - 2) * 32 + dd0) * GN + j) * GN + i;
        x0 = p[0]; x1 = p[NN];
      }
      union { _Float16 h[2]; uint w; } pk;
      pk.h[0] = (_Float16)x0; pk.h[1] = (_Float16)x1;
      *(uint*)(Bt + j * CSTR + dd0) = pk.w;
    }
    __syncthreads();
    const _Float16* arow = ah + (size_t)(mtile * 16 + l15) * 128 + kk * 32;
    v16h a = load_a_frag(arow, half);
#pragma unroll
    for (int t = 0; t < 4; ++t) {
      v16h bf = load_b_frag(Bt, nhalf * 64 + t * 16 + l15, half);
      acc[t] = wmma_f16(a, bf, acc[t]);
    }
    __syncthreads();
  }
  // Epilogue: add rank-1 terms, route to f32 (out0) or f16 (t / oi2).
#pragma unroll
  for (int t = 0; t < 4; ++t) {
#pragma unroll
    for (int v = 0; v < 8; ++v) {
      int s = mtile * 16 + half * 8 + v;          // C/D layout: M = v + 8*half
      int j = nhalf * 64 + t * 16 + l15;          // N = lane&15
      size_t bs = (size_t)b * GC + s;
      float val = acc[t][v] + Ri[bs * GN + i] + Cj[bs * GN + j] + Kc[bs];
      if (i == j) { val += Di[bs * GN + i]; if (dbias) val += dbias[s]; }
      size_t g = (bs * GN + i) * GN + j;
      if (outF) outF[g] = val; else outH[g] = (_Float16)val;
    }
  }
}

// ---------------------------------------------------------------------------
// Batched 128x128x128 matmul + fuse:  u = sw0*out0 + sw1*(Oi2 @ T) + all_bias.
// grid = B*64 blocks (one per (b,s) matrix), 256 threads = 8 waves.
// Wave w owns M-rows [16w, 16w+16) x all 128 columns (8 N-tiles).
// ---------------------------------------------------------------------------
__global__ void rg_bmm_fuse_kernel(const float* __restrict__ out0,
                                   const _Float16* __restrict__ oi2,
                                   const _Float16* __restrict__ th,
                                   const float* __restrict__ sw0p,
                                   const float* __restrict__ sw1p,
                                   const float* __restrict__ abias,
                                   float* __restrict__ uo) {
  __shared__ __align__(16) _Float16 Bt[GN * CSTR];
  int bs = blockIdx.x;
  int s = bs & 63;
  int tid = threadIdx.x, lane = tid & 31, wave = tid >> 5;
  int half = lane >> 4, l15 = lane & 15;
  float sw0 = sw0p[0], sw1 = sw1p[0], ab = abias[s];
  const _Float16* Ab = oi2 + (size_t)bs * NN;
  const _Float16* Bb = th  + (size_t)bs * NN;
  v8f acc[8];
#pragma unroll
  for (int t = 0; t < 8; ++t)
#pragma unroll
    for (int v = 0; v < 8; ++v) acc[t][v] = 0.f;

  for (int kk = 0; kk < 4; ++kk) {
    for (int t = tid; t < 16 * GN; t += 256) {
      int pr = t >> 7;
      int j = t & 127;
      int dd0 = pr * 2;
      const _Float16* p = Bb + (size_t)(kk * 32 + dd0) * GN + j;
      union { _Float16 h[2]; uint w; } pk;
      pk.h[0] = p[0]; pk.h[1] = p[GN];
      *(uint*)(Bt + j * CSTR + dd0) = pk.w;
    }
    __syncthreads();
    const _Float16* arow = Ab + (size_t)(wave * 16 + l15) * GN + kk * 32;
    v16h a = load_a_frag(arow, half);
#pragma unroll
    for (int t = 0; t < 8; ++t) {
      v16h bf = load_b_frag(Bt, t * 16 + l15, half);
      acc[t] = wmma_f16(a, bf, acc[t]);
    }
    __syncthreads();
  }
#pragma unroll
  for (int t = 0; t < 8; ++t) {
#pragma unroll
    for (int v = 0; v < 8; ++v) {
      int i = wave * 16 + half * 8 + v;
      int j = t * 16 + l15;
      size_t g = (size_t)bs * NN + (size_t)i * GN + j;
      uo[g] = sw1 * acc[t][v] + sw0 * out0[g] + ab;
    }
  }
}

// ---------------------------------------------------------------------------
// BatchNorm (training-mode, biased var) over (B,i,j) per channel.
// ---------------------------------------------------------------------------
__global__ void rg_bn_stats_kernel(const float* __restrict__ u,
                                   const float* __restrict__ gamma,
                                   const float* __restrict__ beta,
                                   float* __restrict__ scale, float* __restrict__ shift) {
  int c = blockIdx.x;
  int tid = threadIdx.x;
  float s1 = 0.f, s2 = 0.f;
  for (int b = 0; b < GB; ++b) {
    const float* p = u + ((size_t)b * GC + c) * NN;
    for (int t = tid; t < NN; t += 256) { float x = p[t]; s1 += x; s2 += x * x; }
  }
  __shared__ float a1[256], a2[256];
  a1[tid] = s1; a2[tid] = s2;
  __syncthreads();
  for (int off = 128; off > 0; off >>= 1) {
    if (tid < off) { a1[tid] += a1[tid + off]; a2[tid] += a2[tid + off]; }
    __syncthreads();
  }
  if (tid == 0) {
    const float inv = 1.f / (float)(GB * NN);
    float mean = a1[0] * inv;
    float var = a2[0] * inv - mean * mean;
    float istd = rsqrtf(var + 1e-5f);
    float sc = gamma[c] * istd;
    scale[c] = sc;
    shift[c] = beta[c] - mean * sc;
  }
}

__global__ void rg_bn_apply_kernel(float* __restrict__ u,
                                   const float* __restrict__ scale,
                                   const float* __restrict__ shift) {
  int bc = blockIdx.x;
  int c = bc & 63;
  float sc = scale[c], sh = shift[c];
  float* p = u + (size_t)bc * NN;
  for (int t = threadIdx.x; t < NN; t += 256) p[t] = p[t] * sc + sh;
}

// ---------------------------------------------------------------------------
// feat_extract from per-(b,c) trace & total: o = tr/n @ w1 + b1 + s @ w2;
// o += relu(o) @ w3 ; out (B,128) = / += o.   grid = B, 128 threads.
// ---------------------------------------------------------------------------
__global__ void rg_feat_kernel(const float* __restrict__ tr, const float* __restrict__ tot,
                               const float* __restrict__ w1, const float* __restrict__ b1,
                               const float* __restrict__ w2, const float* __restrict__ w3,
                               int din, float* __restrict__ out, int accumulate) {
  int b = blockIdx.x, f = threadIdx.x;
  const float inv_n = 1.f / (float)GN;
  const float inv_s = 1.f / ((float)GN * (float)(GN - 1));
  float o = b1[f];
  for (int c = 0; c < din; ++c) {
    float t = tr[(size_t)b * din + c];
    float sv = (tot[(size_t)b * din + c] - t) * inv_s;
    o += t * inv_n * w1[(size_t)c * GHF + f] + sv * w2[(size_t)c * GHF + f];
  }
  __shared__ float ro[GHF];
  ro[f] = fmaxf(o, 0.f);
  __syncthreads();
  float o2 = o;
  for (int g = 0; g < GHF; ++g) o2 += ro[g] * w3[(size_t)g * GHF + f];
  if (accumulate) out[b * GHF + f] += o2; else out[b * GHF + f] = o2;
}

// ---------------------------------------------------------------------------
// Head: relu/2, residual MLP, final linear, log-softmax. grid = B, 128 threads.
// ---------------------------------------------------------------------------
__global__ void rg_head_kernel(const float* __restrict__ feat,
                               const float* __restrict__ aw, const float* __restrict__ ab,
                               const float* __restrict__ fw, const float* __restrict__ fb,
                               float* __restrict__ out) {
  int b = blockIdx.x, f = threadIdx.x;
  __shared__ float sh[GHF], sh2[GHF], sz[16];
  float h = fmaxf(feat[b * GHF + f], 0.f) * 0.5f;  // relu / NLAYERS
  sh[f] = h;
  __syncthreads();
  float z = ab[f];
  for (int g = 0; g < GHF; ++g) z += sh[g] * aw[g * GHF + f];
  float h2 = h + fmaxf(z, 0.f);
  sh2[f] = h2;
  __syncthreads();
  if (f < GNC) {
    float zz = fb[f];
    for (int g = 0; g < GHF; ++g) zz += sh2[g] * fw[g * GNC + f];
    sz[f] = zz;
  }
  __syncthreads();
  if (f < GNC) {
    float m = -1e30f;
    for (int c = 0; c < GNC; ++c) m = fmaxf(m, sz[c]);
    float sum = 0.f;
    for (int c = 0; c < GNC; ++c) sum += expf(sz[c] - m);
    out[b * GNC + f] = sz[f] - m - logf(sum);
  }
}

// ---------------------------------------------------------------------------
// Host orchestration.
// d_in order (jax tree-flatten: dict keys sorted, lists in order):
//  0:A  1:after_b 2:after_w 3:bn0.beta 4:bn0.gamma 5:bn1.beta 6:bn1.gamma
//  7:cv0.all_bias 8-10:cv0.coeffs[0..2] 11-13:cv0.diag_bias[0..2] 14-15:cv0.switch
// 16:cv1.all_bias 17-19:cv1.coeffs 20-22:cv1.diag_bias 23-24:cv1.switch
// 25:fe0.b1 26:fe0.w1 27:fe0.w2 28:fe0.w3  29-32:fe1.{b1,w1,w2,w3}
// 33:final_b 34:final_w 35:np.b1 36:np.w1 37:np.w2 38:np.w3
// ---------------------------------------------------------------------------
extern "C" void kernel_launch(void* const* d_in, const int* in_sizes, int n_in,
                              void* d_out, int out_size, void* d_ws, size_t ws_size,
                              hipStream_t stream) {
  const float* A        = (const float*)d_in[0];
  const float* after_b  = (const float*)d_in[1];
  const float* after_w  = (const float*)d_in[2];
  const float* bn_beta[2]  = {(const float*)d_in[3], (const float*)d_in[5]};
  const float* bn_gamma[2] = {(const float*)d_in[4], (const float*)d_in[6]};
  const float* cv_ab[2]    = {(const float*)d_in[7],  (const float*)d_in[16]};
  const float* cv_co[2][3] = {{(const float*)d_in[8],  (const float*)d_in[9],  (const float*)d_in[10]},
                              {(const float*)d_in[17], (const float*)d_in[18], (const float*)d_in[19]}};
  const float* cv_db[2][3] = {{(const float*)d_in[11], (const float*)d_in[12], (const float*)d_in[13]},
                              {(const float*)d_in[20], (const float*)d_in[21], (const float*)d_in[22]}};
  const float* cv_sw[2][2] = {{(const float*)d_in[14], (const float*)d_in[15]},
                              {(const float*)d_in[23], (const float*)d_in[24]}};
  const float* fe_b1[2] = {(const float*)d_in[25], (const float*)d_in[29]};
  const float* fe_w1[2] = {(const float*)d_in[26], (const float*)d_in[30]};
  const float* fe_w2[2] = {(const float*)d_in[27], (const float*)d_in[31]};
  const float* fe_w3[2] = {(const float*)d_in[28], (const float*)d_in[32]};
  const float* final_b = (const float*)d_in[33];
  const float* final_w = (const float*)d_in[34];
  const float* np_b1 = (const float*)d_in[35];
  const float* np_w1 = (const float*)d_in[36];
  const float* np_w2 = (const float*)d_in[37];
  const float* np_w3 = (const float*)d_in[38];

  // ---- carve workspace -----------------------------------------------------
  char* W = (char*)d_ws;
  auto alloc = [&](size_t bytes) {
    char* p = W;
    W += (bytes + 255) & ~(size_t)255;
    return p;
  };
  const size_t BCnn = (size_t)GB * GC * NN;
  float*    u     = (float*)alloc(BCnn * sizeof(float));       // feature map
  float*    out0  = (float*)alloc(BCnn * sizeof(float));       // idx0 result (f32)
  _Float16* th    = (_Float16*)alloc(BCnn * sizeof(_Float16)); // idx1 result (f16)
  _Float16* oi2h  = (_Float16*)alloc(BCnn * sizeof(_Float16)); // idx2 result (f16)
  _Float16* ah    = (_Float16*)alloc((size_t)GC * 128 * sizeof(_Float16)); // packed A'
  float* rows = (float*)alloc((size_t)GB * GC * GN * sizeof(float));
  float* cols = (float*)alloc((size_t)GB * GC * GN * sizeof(float));
  float* diag = (float*)alloc((size_t)GB * GC * GN * sizeof(float));
  float* sd   = (float*)alloc((size_t)GB * GC * sizeof(float));
  float* tot  = (float*)alloc((size_t)GB * GC * sizeof(float));
  float* Ri   = (float*)alloc((size_t)GB * GC * GN * sizeof(float));
  float* Cj   = (float*)alloc((size_t)GB * GC * GN * sizeof(float));
  float* Di   = (float*)alloc((size_t)GB * GC * GN * sizeof(float));
  float* Kc   = (float*)alloc((size_t)GB * GC * sizeof(float));
  float* rowsA = (float*)alloc((size_t)GB * GN * sizeof(float));
  float* colsA = (float*)alloc((size_t)GB * GN * sizeof(float));
  float* diagA = (float*)alloc((size_t)GB * GN * sizeof(float));
  float* sdA   = (float*)alloc((size_t)GB * sizeof(float));
  float* totA  = (float*)alloc((size_t)GB * sizeof(float));
  float* bn_scale = (float*)alloc(GC * sizeof(float));
  float* bn_shift = (float*)alloc(GC * sizeof(float));
  float* feat = (float*)alloc((size_t)GB * GHF * sizeof(float));
  (void)ws_size; (void)n_in; (void)in_sizes; (void)out_size;

  // ---- no_prop branch on A -------------------------------------------------
  rg_stats_kernel<<<GB, GN, 0, stream>>>(A, rowsA, colsA, diagA, sdA, totA);
  rg_feat_kernel<<<GB, GHF, 0, stream>>>(sdA, totA, np_w1, np_b1, np_w2, np_w3,
                                         1, feat, 0);

  // ---- layer 1 (din = 1) ---------------------------------------------------
  // idx0 -> out0 (f32), idx1 -> th (f16), idx2 -> oi2h (f16, no diag_bias)
  rg_vec_kernel<<<GB * GC, GN, 0, stream>>>(cv_co[0][0], 1, rowsA, colsA, diagA,
                                            sdA, totA, Ri, Cj, Di, Kc);
  rg_mix1_kernel<<<GB * GC, 256, 0, stream>>>(A, cv_co[0][0], Ri, Cj, Di, Kc,
                                              cv_db[0][0], out0, nullptr);
  rg_vec_kernel<<<GB * GC, GN, 0, stream>>>(cv_co[0][1], 1, rowsA, colsA, diagA,
                                            sdA, totA, Ri, Cj, Di, Kc);
  rg_mix1_kernel<<<GB * GC, 256, 0, stream>>>(A, cv_co[0][1], Ri, Cj, Di, Kc,
                                              cv_db[0][1], nullptr, th);
  rg_vec_kernel<<<GB * GC, GN, 0, stream>>>(cv_co[0][2], 1, rowsA, colsA, diagA,
                                            sdA, totA, Ri, Cj, Di, Kc);
  rg_mix1_kernel<<<GB * GC, 256, 0, stream>>>(A, cv_co[0][2], Ri, Cj, Di, Kc,
                                              nullptr, nullptr, oi2h);
  rg_bmm_fuse_kernel<<<GB * GC, 256, 0, stream>>>(out0, oi2h, th, cv_sw[0][0],
                                                  cv_sw[0][1], cv_ab[0], u);
  rg_bn_stats_kernel<<<GC, 256, 0, stream>>>(u, bn_gamma[0], bn_beta[0],
                                             bn_scale, bn_shift);
  rg_bn_apply_kernel<<<GB * GC, 256, 0, stream>>>(u, bn_scale, bn_shift);
  rg_stats_kernel<<<GB * GC, GN, 0, stream>>>(u, rows, cols, diag, sd, tot);
  rg_feat_kernel<<<GB, GHF, 0, stream>>>(sd, tot, fe_w1[0], fe_b1[0], fe_w2[0],
                                         fe_w3[0], GC, feat, 1);

  // ---- layer 2 (din = 64, WMMA channel-mix) --------------------------------
  for (int idx = 0; idx < 3; ++idx) {
    rg_packA_kernel<<<1, 256, 0, stream>>>(cv_co[1][idx], ah);
    rg_vec_kernel<<<GB * GC, GN, 0, stream>>>(cv_co[1][idx], GC, rows, cols, diag,
                                              sd, tot, Ri, Cj, Di, Kc);
    if (idx == 0)
      rg_mix64_kernel<<<GB * GN, 256, 0, stream>>>(u, ah, Ri, Cj, Di, Kc,
                                                   cv_db[1][0], out0, nullptr);
    else if (idx == 1)
      rg_mix64_kernel<<<GB * GN, 256, 0, stream>>>(u, ah, Ri, Cj, Di, Kc,
                                                   cv_db[1][1], nullptr, th);
    else
      rg_mix64_kernel<<<GB * GN, 256, 0, stream>>>(u, ah, Ri, Cj, Di, Kc,
                                                   nullptr, nullptr, oi2h);
  }
  rg_bmm_fuse_kernel<<<GB * GC, 256, 0, stream>>>(out0, oi2h, th, cv_sw[1][0],
                                                  cv_sw[1][1], cv_ab[1], u);
  rg_bn_stats_kernel<<<GC, 256, 0, stream>>>(u, bn_gamma[1], bn_beta[1],
                                             bn_scale, bn_shift);
  rg_bn_apply_kernel<<<GB * GC, 256, 0, stream>>>(u, bn_scale, bn_shift);
  rg_stats_kernel<<<GB * GC, GN, 0, stream>>>(u, rows, cols, diag, sd, tot);
  rg_feat_kernel<<<GB, GHF, 0, stream>>>(sd, tot, fe_w1[1], fe_b1[1], fe_w2[1],
                                         fe_w3[1], GC, feat, 1);

  // ---- head ----------------------------------------------------------------
  rg_head_kernel<<<GB, GHF, 0, stream>>>(feat, after_w, after_b, final_w,
                                         final_b, (float*)d_out);
}
